// IsoEncoderLayer_66941360275957
// MI455X (gfx1250) — matrix operand order repaired and checked by reading, not compiled, not run on hardware
//
#include <hip/hip_runtime.h>
#include <hip/hip_bf16.h>
#include <cstdint>

// Problem constants (from reference): B=8, S=2048, D=512, E=8, H=8
#define BB 8
#define SS 2048
#define DD 512
#define EE 8
#define HH 8
#define MT 64            // rows (sequence positions) per workgroup strip
#define LDP 520          // padded LDS row pitch in bf16 elems (1040B, breaks bank alignment)
#define NSTRIP (SS / MT) // 32 strips per (b,e)

typedef __attribute__((ext_vector_type(16))) __bf16 bf16x16;
typedef __attribute__((ext_vector_type(8)))  float  v8f;
typedef __attribute__((ext_vector_type(4)))  unsigned int v4u;
typedef __attribute__((ext_vector_type(8)))  int v8i;
typedef __attribute__((ext_vector_type(4)))  int v4i;

union BF16Frag { bf16x16 v; uint4 q[2]; };
union BFPack2  { __bf16 h[2]; unsigned int u; };

// ---- monotone float <-> int encoding so we can use integer atomicMax ----
__device__ __forceinline__ int enc_f(float f) {
    int b = __float_as_int(f);
    return b >= 0 ? b : (b ^ 0x7fffffff);
}
__device__ __forceinline__ float dec_f(int k) {
    int b = k >= 0 ? k : (k ^ 0x7fffffff);
    return __int_as_float(b);
}

__device__ __forceinline__ float elu_f(float t) {
    return t > 0.0f ? t : (__expf(t) - 1.0f);
}

// ---- WMMA fragment loads per CDNA5 ISA layouts (wave32) ----
// A (16x32 bf16): lane<16 -> row m0+lane, K {k0..k0+7} U {k0+16..k0+23}
//                 lane>=16 -> row m0+lane-16, K shifted by +8
__device__ __forceinline__ bf16x16 load_A(const __bf16* lds, int m0, int k0,
                                          int ldk, int lane) {
    int half = lane >> 4;
    int row  = m0 + (lane & 15);
    const __bf16* p = lds + row * ldk + k0 + half * 8;
    BF16Frag r;
    r.q[0] = *(const uint4*)p;         // K = k0+half*8 .. +7
    r.q[1] = *(const uint4*)(p + 16);  // K = k0+16+half*8 .. +7
    return r.v;
}

// B (32x16 bf16), weights row-major [n][k] (k contiguous):
// lane group 0-15:  N = n0+lane,    K {k0..k0+7} U {k0+16..k0+23}
// lane group 16-31: N = n0+lane-16, K shifted by +8
__device__ __forceinline__ bf16x16 load_B(const __bf16* W, int n0, int k0,
                                          int ldk, int lane) {
    const __bf16* p = W + (size_t)(n0 + (lane & 15)) * ldk + k0 + (lane >> 4) * 8;
    __builtin_prefetch(p + 64, 0, 1);  // next K-steps -> global_prefetch_b8
    BF16Frag r;
    r.q[0] = *(const uint4*)p;
    r.q[1] = *(const uint4*)(p + 16);
    return r.v;
}

__device__ __forceinline__ v8f wmma_bb(bf16x16 a, bf16x16 b, v8f c) {
    return __builtin_amdgcn_wmma_f32_16x16x32_bf16(
        /*neg_a=*/false, a, /*neg_b=*/false, b,
        /*c_mod=*/(short)0, c, /*reuse_a=*/false, /*reuse_b=*/false);
}

// 4 M-tiles x 2 N-tiles register tile per wave: each pair of B fragments
// feeds 8 WMMAs (4x less global weight traffic than 1Mx2N tiling).
__device__ __forceinline__ void gemm_4x2(const __bf16* As, int lda,
                                         const __bf16* Bm, int ldb,
                                         int n0a, int n0b, int lane,
                                         v8f acc[4][2]) {
    for (int k0 = 0; k0 < DD; k0 += 32) {
        bf16x16 b0 = load_B(Bm, n0a, k0, ldb, lane);
        bf16x16 b1 = load_B(Bm, n0b, k0, ldb, lane);
#pragma unroll
        for (int mt = 0; mt < 4; ++mt) {
            bf16x16 a = load_A(As, mt * 16, k0, lda, lane);
            acc[mt][0] = wmma_bb(a, b0, acc[mt][0]);
            acc[mt][1] = wmma_bb(a, b1, acc[mt][1]);
        }
    }
}

// ---- gfx1250 async global->LDS copy (ASYNCcnt-tracked, no VGPR bounce) ----
__device__ __forceinline__ void async_copy_b128(void* lds_dst, const void* gsrc) {
    unsigned lds = (unsigned)(uintptr_t)lds_dst;  // flat-shared low 32 bits == LDS offset
    unsigned long long ga = (unsigned long long)(uintptr_t)gsrc;
    asm volatile("global_load_async_to_lds_b128 %0, %1, off"
                 :: "v"(lds), "v"(ga) : "memory");
}
__device__ __forceinline__ void wait_async0() {
    asm volatile("s_wait_asynccnt 0x0" ::: "memory");
}

// ---- Tensor Data Mover: DMA a 64-row x 1024B tile into LDS with a 16B
// per-row pad so the landed pitch is LDP bf16 elems. D# per ISA ch.8:
// 2D tile, data_size=4B, tile 256x64 dwords, pad 4 dwords every 256. ----
__device__ __forceinline__ void tdm_load_strip(const void* gsrc, void* lds_dst) {
    unsigned long long ga = (unsigned long long)(uintptr_t)gsrc;
    v4u g0;
    g0[0] = 1u;                                     // count=1 valid descriptor
    g0[1] = (unsigned)(uintptr_t)lds_dst;           // lds_addr (byte offset)
    g0[2] = (unsigned)(ga & 0xffffffffu);           // global_addr[31:0]
    g0[3] = (unsigned)((ga >> 32) & 0x1ffffffu) | (2u << 30);  // addr[56:32] | type=2
    v8i g1;
    g1[0] = (int)((2u << 16)          // data_size = 4 bytes
                | (1u << 20)          // pad_enable
                | (7u << 22)          // pad_interval: every 256 dwords
                | (3u << 25));        // pad_amount: 4 dwords (16B) -> pitch 1040B
    g1[1] = (int)(256u << 16);        // tensor_dim0 = 256 dwords  (bits 79:48 lo)
    g1[2] = (int)(64u << 16);         // tensor_dim1 = 64 rows     (bits 111:80 lo)
    g1[3] = (int)(256u << 16);        // tile_dim0 = 256 dwords    (bits 127:112)
    g1[4] = (int)64u;                 // tile_dim1 = 64 rows       (bits 143:128)
    g1[5] = (int)256u;                // tensor_dim0_stride = 256 dwords (lo32)
    g1[6] = (int)(256u << 16);        // tensor_dim1_stride lo16   (bits 223:208)
    g1[7] = 0;
    v4i g2 = {0, 0, 0, 0};
    v4i g3 = {0, 0, 0, 0};
#if __clang_major__ >= 23
    v8i g4 = {0, 0, 0, 0, 0, 0, 0, 0};
    __builtin_amdgcn_tensor_load_to_lds(g0, g1, g2, g3, g4, 0);
#else
    __builtin_amdgcn_tensor_load_to_lds(g0, g1, g2, g3, 0);
#endif
    __builtin_amdgcn_s_wait_tensorcnt((short)0);
}

// =====================================================================
// K0: conversions + gates + pooled init
// =====================================================================
__global__ void prep_kernel(const float* __restrict__ x, const float* __restrict__ WU,
                            const float* __restrict__ W1, const float* __restrict__ W2,
                            const float* __restrict__ rk,
                            __bf16* __restrict__ xh, __bf16* __restrict__ wuh,
                            __bf16* __restrict__ w1h, __bf16* __restrict__ w2h,
                            float* __restrict__ ab, int* __restrict__ pooled) {
    int i = blockIdx.x * blockDim.x + threadIdx.x;
    int stride = gridDim.x * blockDim.x;
    const int NX = BB * SS * DD;
    for (int idx = i; idx < NX; idx += stride) {
        xh[idx] = (__bf16)x[idx];
        if (idx < EE * DD * DD) wuh[idx] = (__bf16)WU[idx];
        if (idx < DD * DD) { w1h[idx] = (__bf16)W1[idx]; w2h[idx] = (__bf16)W2[idx]; }
        if (idx < EE * DD) {
            int e = idx / DD, d = idx % DD, h = d % HH;
            float r = rk[e * HH + h];
            float av = r <= 0.25f ? 1.5f : (r <= 0.5f ? 1.5f : (r <= 0.75f ? 0.0f : 1.0f));
            float bv = r <= 0.25f ? 1.0f : (r <= 0.5f ? -1.0f : (r <= 0.75f ? 4.0f : 0.0f));
            ab[idx] = av;
            ab[EE * DD + idx] = bv;
        }
        if (idx < BB * EE * DD) pooled[idx] = (int)0x80000000;
    }
}

// =====================================================================
// K1: eig_v^T[e][f][d] = sum_k W_V[e,f,k] * eig[d,k] + b_V[e,f]  (bf16 out)
// =====================================================================
__global__ void eigv_kernel(const float* __restrict__ eig, const float* __restrict__ WV,
                            const float* __restrict__ bV, __bf16* __restrict__ evh) {
    int idx = blockIdx.x * blockDim.x + threadIdx.x;
    if (idx >= EE * DD * DD) return;
    int e = idx / (DD * DD);
    int f = (idx / DD) % DD;
    int d = idx % DD;
    const float4* wr = (const float4*)(WV + ((size_t)e * DD + f) * DD);
    const float4* er = (const float4*)(eig + (size_t)d * DD);
    float s = 0.0f;
    for (int k = 0; k < DD / 4; ++k) {
        float4 a = wr[k], b = er[k];
        s += a.x * b.x + a.y * b.y + a.z * b.z + a.w * b.w;
    }
    evh[idx] = (__bf16)(s + bV[e * DD + f]);
}

// =====================================================================
// K2: fused per-(b,e) expert. TDM stages the x strip; two chained WMMA
// GEMMs; epilogue computes out = a*x + b*elu(.)*rw, bf16 store + pooled max.
// grid = B*E*NSTRIP blocks of 256 threads (8 waves)
// =====================================================================
__global__ __launch_bounds__(256) void expert_kernel(
    const float* __restrict__ x, const __bf16* __restrict__ xh,
    const __bf16* __restrict__ wuh, const __bf16* __restrict__ evh,
    const float* __restrict__ bU, const float* __restrict__ ab,
    const float* __restrict__ rwp,
    __bf16* __restrict__ outb, int* __restrict__ pooled) {
    extern __shared__ __align__(16) char smem[];
    __bf16* xs = (__bf16*)smem;       // MT x LDP (x strip, bf16)
    __bf16* us = xs + MT * LDP;       // MT x LDP (ux strip, bf16)

    const int bid   = blockIdx.x;
    const int strip = bid % NSTRIP;
    const int e     = (bid / NSTRIP) % EE;
    const int b     = bid / (NSTRIP * EE);
    const int s0    = strip * MT;
    const int tid   = threadIdx.x;
    const int lane  = tid & 31;
    const int wave  = tid >> 5;
    const float rwv = rwp[0];

    // Tensor Data Mover: one wave DMAs the whole strip (EXEC-independent,
    // TENSORcnt-tracked); landed pitch is LDP via D# padding.
    if (wave == 0) {
        tdm_load_strip(xh + ((size_t)b * SS + s0) * DD, xs);
    }
    __syncthreads();

    const __bf16* WU = wuh + (size_t)e * DD * DD;
    const int half = lane >> 4;
    const int nlo  = lane & 15;

    // ---- GEMM1: ux = x @ WU^T + bU  -> us (bf16 in LDS) ----
    for (int jp = 0; jp < 2; ++jp) {
        int n0a = (wave + 16 * jp) * 16;
        int n0b = n0a + 128;
        v8f acc[4][2] = {};
        gemm_4x2(xs, LDP, WU, DD, n0a, n0b, lane, acc);
        float bu0 = bU[e * DD + n0a + nlo];
        float bu1 = bU[e * DD + n0b + nlo];
        for (int mt = 0; mt < 4; ++mt)
            for (int r = 0; r < 8; ++r) {
                int m = mt * 16 + r + half * 8;
                us[m * LDP + n0a + nlo] = (__bf16)(acc[mt][0][r] + bu0);
                us[m * LDP + n0b + nlo] = (__bf16)(acc[mt][1][r] + bu1);
            }
    }
    __syncthreads();

    // ---- GEMM2 + epilogue ----
    const __bf16* EV = evh + (size_t)e * DD * DD;
    __bf16* outp = outb + (((size_t)b * EE + e) * SS + s0) * DD;
    int* poolp = pooled + ((size_t)b * EE + e) * DD;
    for (int jp = 0; jp < 2; ++jp) {
        int n0a = (wave + 16 * jp) * 16;
        int n0b = n0a + 128;
        v8f acc[4][2] = {};
        gemm_4x2(us, LDP, EV, DD, n0a, n0b, lane, acc);
        int na = n0a + nlo, nb = n0b + nlo;
        float aa  = ab[e * DD + na], ba = ab[EE * DD + e * DD + na];
        float aa2 = ab[e * DD + nb], bb2 = ab[EE * DD + e * DD + nb];
        float mxa = -3.4e38f, mxb = -3.4e38f;
        for (int mt = 0; mt < 4; ++mt)
            for (int r = 0; r < 8; ++r) {
                int m = mt * 16 + r + half * 8;
                const float* xrow = x + ((size_t)b * SS + s0 + m) * DD;
                float x20 = elu_f(acc[mt][0][r]) * rwv;
                float o0  = aa * xrow[na] + ba * x20;
                outp[(size_t)m * DD + na] = (__bf16)o0;
                mxa = fmaxf(mxa, o0);
                float x21 = elu_f(acc[mt][1][r]) * rwv;
                float o1  = aa2 * xrow[nb] + bb2 * x21;
                outp[(size_t)m * DD + nb] = (__bf16)o1;
                mxb = fmaxf(mxb, o1);
            }
        // combine the two M-halves (same N set) across lane groups
        mxa = fmaxf(mxa, __shfl_xor(mxa, 16));
        mxb = fmaxf(mxb, __shfl_xor(mxb, 16));
        if (half == 0) {   // one atomic per column per strip
            atomicMax(poolp + na, enc_f(mxa));
            atomicMax(poolp + nb, enc_f(mxb));
        }
    }
}

// =====================================================================
// K3: logits = pooled . W_c + b_c ; ew = softmax over experts
// =====================================================================
__global__ void softmax_kernel(const int* __restrict__ pooled, const float* __restrict__ Wc,
                               const float* __restrict__ bcp, float* __restrict__ ew) {
    __shared__ float lg[EE];
    int b = blockIdx.x;
    int tid = threadIdx.x, lane = tid & 31, wave = tid >> 5;
    const int* pp = pooled + ((size_t)b * EE + wave) * DD;
    float s = 0.0f;
    for (int d = lane; d < DD; d += 32) s += dec_f(pp[d]) * Wc[d];
    for (int off = 16; off > 0; off >>= 1) s += __shfl_xor(s, off);
    if (lane == 0) lg[wave] = s + bcp[0];
    __syncthreads();
    if (tid == 0) {
        float m = lg[0];
        for (int i = 1; i < EE; ++i) m = fmaxf(m, lg[i]);
        float den = 0.0f, v[EE];
        for (int i = 0; i < EE; ++i) { v[i] = __expf(lg[i] - m); den += v[i]; }
        for (int i = 0; i < EE; ++i) ew[b * EE + i] = v[i] / den;
    }
}

// =====================================================================
// K4: comb[b,s,d] = sum_e ew[b,e] * out[b,e,s,d]  (fp32 + bf16 copies)
// =====================================================================
__global__ void combine_kernel(const __bf16* __restrict__ outb, const float* __restrict__ ew,
                               float* __restrict__ comb, __bf16* __restrict__ combh) {
    int i = blockIdx.x * blockDim.x + threadIdx.x;
    const int per_b = SS * DD / 8;
    if (i >= BB * per_b) return;
    int b = i / per_b;
    int r = i % per_b;
    float acc[8] = {0, 0, 0, 0, 0, 0, 0, 0};
    for (int e = 0; e < EE; ++e) {
        float w = ew[b * EE + e];
        uint4 q = *(const uint4*)(outb + ((size_t)b * EE + e) * SS * DD + (size_t)r * 8);
        unsigned int u[4] = {q.x, q.y, q.z, q.w};
        for (int j = 0; j < 4; ++j) {
            float lo = __uint_as_float(u[j] << 16);
            float hi = __uint_as_float(u[j] & 0xffff0000u);
            acc[2 * j]     += w * lo;
            acc[2 * j + 1] += w * hi;
        }
    }
    float* cp = comb + (size_t)b * SS * DD + (size_t)r * 8;
    *(float4*)(cp)     = make_float4(acc[0], acc[1], acc[2], acc[3]);
    *(float4*)(cp + 4) = make_float4(acc[4], acc[5], acc[6], acc[7]);
    uint4 oq;
    BFPack2 t;
    t.h[0] = (__bf16)acc[0]; t.h[1] = (__bf16)acc[1]; oq.x = t.u;
    t.h[0] = (__bf16)acc[2]; t.h[1] = (__bf16)acc[3]; oq.y = t.u;
    t.h[0] = (__bf16)acc[4]; t.h[1] = (__bf16)acc[5]; oq.z = t.u;
    t.h[0] = (__bf16)acc[6]; t.h[1] = (__bf16)acc[7]; oq.w = t.u;
    *(uint4*)(combh + (size_t)b * SS * DD + (size_t)r * 8) = oq;
}

// =====================================================================
// K5: h = elu(comb@W1^T + b1) ; ff = elu(h@W2^T + b2) ; out = comb + rw*ff
// async-to-LDS staging (ASYNCcnt), then two WMMA GEMMs.
// =====================================================================
__global__ __launch_bounds__(256) void ffn_kernel(
    const float* __restrict__ comb, const __bf16* __restrict__ combh,
    const __bf16* __restrict__ w1h, const __bf16* __restrict__ w2h,
    const float* __restrict__ b1, const float* __restrict__ b2,
    const float* __restrict__ rwp, float* __restrict__ out) {
    extern __shared__ __align__(16) char smem[];
    __bf16* cs = (__bf16*)smem;
    __bf16* hs = cs + MT * LDP;

    int bid = blockIdx.x;
    int strip = bid % NSTRIP, b = bid / NSTRIP;
    int s0 = strip * MT;
    int tid = threadIdx.x, lane = tid & 31, wave = tid >> 5;
    float rwv = rwp[0];

    // async global->LDS staging: no VGPR bounce, tracked on ASYNCcnt
    {
        const uint4* src = (const uint4*)(combh + ((size_t)b * SS + s0) * DD);
        for (int i = tid; i < MT * DD / 8; i += 256) {
            int row = i >> 6, c8 = i & 63;
            async_copy_b128(cs + row * LDP + c8 * 8, src + row * 64 + c8);
        }
        wait_async0();
    }
    __syncthreads();

    int half = lane >> 4, nlo = lane & 15;

    for (int jp = 0; jp < 2; ++jp) {
        int n0a = (wave + 16 * jp) * 16, n0b = n0a + 128;
        v8f acc[4][2] = {};
        gemm_4x2(cs, LDP, w1h, DD, n0a, n0b, lane, acc);
        float c0 = b1[n0a + nlo], c1 = b1[n0b + nlo];
        for (int mt = 0; mt < 4; ++mt)
            for (int r = 0; r < 8; ++r) {
                int m = mt * 16 + r + half * 8;
                hs[m * LDP + n0a + nlo] = (__bf16)elu_f(acc[mt][0][r] + c0);
                hs[m * LDP + n0b + nlo] = (__bf16)elu_f(acc[mt][1][r] + c1);
            }
    }
    __syncthreads();

    const float* combp = comb + ((size_t)b * SS + s0) * DD;
    float* outp = out + ((size_t)b * SS + s0) * DD;
    for (int jp = 0; jp < 2; ++jp) {
        int n0a = (wave + 16 * jp) * 16, n0b = n0a + 128;
        v8f acc[4][2] = {};
        gemm_4x2(hs, LDP, w2h, DD, n0a, n0b, lane, acc);
        float c0 = b2[n0a + nlo], c1 = b2[n0b + nlo];
        for (int mt = 0; mt < 4; ++mt)
            for (int r = 0; r < 8; ++r) {
                int m = mt * 16 + r + half * 8;
                float f0 = elu_f(acc[mt][0][r] + c0);
                float f1 = elu_f(acc[mt][1][r] + c1);
                outp[(size_t)m * DD + n0a + nlo] = combp[(size_t)m * DD + n0a + nlo] + rwv * f0;
                outp[(size_t)m * DD + n0b + nlo] = combp[(size_t)m * DD + n0b + nlo] + rwv * f1;
            }
    }
}

// =====================================================================
extern "C" void kernel_launch(void* const* d_in, const int* in_sizes, int n_in,
                              void* d_out, int out_size, void* d_ws, size_t ws_size,
                              hipStream_t stream) {
    (void)in_sizes; (void)n_in; (void)out_size; (void)ws_size;
    const float* x   = (const float*)d_in[0];
    const float* eig = (const float*)d_in[1];
    const float* rw  = (const float*)d_in[2];
    const float* rk  = (const float*)d_in[3];
    const float* WU  = (const float*)d_in[4];
    const float* bU  = (const float*)d_in[5];
    const float* WV  = (const float*)d_in[6];
    const float* bV  = (const float*)d_in[7];
    const float* Wc  = (const float*)d_in[8];
    const float* bc  = (const float*)d_in[9];
    const float* W1  = (const float*)d_in[10];
    const float* b1  = (const float*)d_in[11];
    const float* W2  = (const float*)d_in[12];
    const float* b2  = (const float*)d_in[13];
    float* out = (float*)d_out;

    char* ws = (char*)d_ws;
    size_t off = 0;
    auto alloc = [&](size_t bytes) -> char* {
        char* p = ws + off;
        off = (off + bytes + 255) & ~(size_t)255;
        return p;
    };
    __bf16* xh    = (__bf16*)alloc((size_t)BB * SS * DD * 2);   // x in bf16
    __bf16* wuh   = (__bf16*)alloc((size_t)EE * DD * DD * 2);   // W_U bf16
    __bf16* evh   = (__bf16*)alloc((size_t)EE * DD * DD * 2);   // eig_v^T bf16
    __bf16* w1h   = (__bf16*)alloc((size_t)DD * DD * 2);
    __bf16* w2h   = (__bf16*)alloc((size_t)DD * DD * 2);
    float*  ab    = (float*) alloc((size_t)2 * EE * DD * 4);    // a_d, b_d
    int*    pool  = (int*)   alloc((size_t)BB * EE * DD * 4);   // encoded max
    float*  ew    = (float*) alloc((size_t)BB * EE * 4);
    __bf16* outb  = (__bf16*)alloc((size_t)BB * EE * SS * DD * 2);
    float*  comb  = (float*) alloc((size_t)BB * SS * DD * 4);
    __bf16* combh = (__bf16*)alloc((size_t)BB * SS * DD * 2);

    const size_t lds = (size_t)2 * MT * LDP * sizeof(__bf16);   // ~130 KB

    prep_kernel<<<(BB * SS * DD + 255) / 256, 256, 0, stream>>>(
        x, WU, W1, W2, rk, xh, wuh, w1h, w2h, ab, pool);
    eigv_kernel<<<(EE * DD * DD + 255) / 256, 256, 0, stream>>>(eig, WV, bV, evh);
    expert_kernel<<<BB * EE * NSTRIP, 256, lds, stream>>>(
        x, xh, wuh, evh, bU, ab, rw, outb, pool);
    softmax_kernel<<<BB, 256, 0, stream>>>(pool, Wc, bc, ew);
    combine_kernel<<<(BB * SS * DD / 8 + 255) / 256, 256, 0, stream>>>(
        outb, ew, comb, combh);
    ffn_kernel<<<BB * NSTRIP, 256, lds, stream>>>(
        comb, combh, w1h, w2h, b1, b2, rw, out);
}
